// MLPAttention_7687991459887
// MI455X (gfx1250) — compile-verified, matrix operand order checked by reading, not run on hardware
//
#include <hip/hip_runtime.h>
#include <hip/hip_bf16.h>
#include <math.h>

typedef __attribute__((ext_vector_type(16))) _Float16 v16h;
typedef __attribute__((ext_vector_type(8)))  _Float16 v8h;
typedef __attribute__((ext_vector_type(4)))  _Float16 v4h;
typedef __attribute__((ext_vector_type(8)))  float    v8f;

#define WMMA_F16(a, b, c) \
  __builtin_amdgcn_wmma_f32_16x16x32_f16(false, (a), false, (b), (short)0, (c), false, false)

// Load a 16x32 f16 WMMA fragment for this lane.
// Per ISA 16-bit A/B layout: lane L (hi = L>>4) holds K groups
// [hi*8, hi*8+8) in v[0..7] and [16+hi*8, 16+hi*8+8) in v[8..15],
// i.e. two contiguous 16-byte chunks at +0 and +16 halfs past (base + hi*8).
template <typename P>
static __device__ __forceinline__ v16h load_frag(const P* p) {
  v16h r;
  ((v8h*)&r)[0] = *(const v8h*)(p);
  ((v8h*)&r)[1] = *(const v8h*)(p + 16);
  return r;
}

// ---------------------------------------------------------------------------
// fp32 -> f16 conversion (vectorized x4)
// ---------------------------------------------------------------------------
__global__ __launch_bounds__(256) void cvt_f16_kernel(const float* __restrict__ src,
                                                      _Float16* __restrict__ dst, int n4) {
  int i = blockIdx.x * 256 + threadIdx.x;
  if (i < n4) {
    float4 f = *(const float4*)(src + (size_t)i * 4);
    v4h h;
    h[0] = (_Float16)f.x; h[1] = (_Float16)f.y;
    h[2] = (_Float16)f.z; h[3] = (_Float16)f.w;
    *(v4h*)(dst + (size_t)i * 4) = h;
  }
}

// ---------------------------------------------------------------------------
// M=32 GEMM:  out[m,e] = sum_k A[m,k] * W[e,k]   (e in [0,1024), m in [0,32))
// mode 0: out = acc + bias[e]      mode 1: out = tanh(acc)
// One block, 8 waves; wave w owns columns [w*128, w*128+128).
// ---------------------------------------------------------------------------
__global__ __launch_bounds__(256) void gemm32_kernel(const float* __restrict__ A,
                                                     const _Float16* __restrict__ W,
                                                     const float* __restrict__ bias,
                                                     float* __restrict__ out,
                                                     int K, int mode) {
  constexpr int KC = 512, LDK = KC + 8;
  __shared__ __align__(16) _Float16 Ash[32 * LDK];
  const int tid = threadIdx.x;
  const int wave = tid >> 5, lane = tid & 31;
  const int Lm = lane & 15, hi = lane >> 4;
  const int nBase = wave * 128;

  v8f acc[2][8] = {};

  for (int k0 = 0; k0 < K; k0 += KC) {
    // stage 32 x KC chunk of A into LDS as f16
    for (int idx = tid; idx < 32 * (KC / 4); idx += 256) {
      const int row = idx >> 7;            // KC/4 == 128
      const int c4  = (idx & 127) << 2;
      float4 f = *(const float4*)(A + (size_t)row * K + k0 + c4);
      v4h h;
      h[0] = (_Float16)f.x; h[1] = (_Float16)f.y;
      h[2] = (_Float16)f.z; h[3] = (_Float16)f.w;
      *(v4h*)&Ash[row * LDK + c4] = h;
    }
    __syncthreads();

    for (int kc = 0; kc < KC; kc += 32) {
      const _Float16* ap = &Ash[Lm * LDK + kc + hi * 8];
      const v16h a0 = load_frag(ap);
      const v16h a1 = load_frag(ap + 16 * LDK);
      const _Float16* wp = W + (size_t)(nBase + Lm) * K + k0 + kc + hi * 8;
#pragma unroll
      for (int j = 0; j < 8; ++j) {
        const v16h b = load_frag(wp + (size_t)j * 16 * K);
        acc[0][j] = WMMA_F16(a0, b, acc[0][j]);
        acc[1][j] = WMMA_F16(a1, b, acc[1][j]);
      }
    }
    __syncthreads();
  }

  // epilogue: C layout -> lane holds (m = ms*16 + hi*8 + r, e = nBase + j*16 + Lm)
#pragma unroll
  for (int ms = 0; ms < 2; ++ms)
#pragma unroll
    for (int j = 0; j < 8; ++j) {
      const int e = nBase + j * 16 + Lm;
      const float bv = (mode == 0) ? bias[e] : 0.f;
#pragma unroll
      for (int r = 0; r < 8; ++r) {
        const int m = ms * 16 + hi * 8 + r;
        float v = acc[ms][j][r] + bv;
        if (mode == 1) v = tanhf(v);
        out[m * 1024 + e] = v;
      }
    }
}

// ---------------------------------------------------------------------------
// Fused: scores[row] = sum_e W_v[e] * tanh(t[b,e] + sum_d context[row,d]*W_c[e,d])
// Block = 32 context rows (all within one batch b), 8 waves, full e=1024 sweep.
// Never materializes the (B,S,D) "source" tensor.
// ---------------------------------------------------------------------------
__global__ __launch_bounds__(256) void attn_scores_kernel(const float* __restrict__ context,
                                                          const _Float16* __restrict__ Wc,
                                                          const float* __restrict__ t_ws,
                                                          const float* __restrict__ Wv,
                                                          float* __restrict__ scores) {
  constexpr int K = 1024, KC = 512, LDK = KC + 8;
  __shared__ __align__(16) _Float16 Ash[32 * LDK];
  __shared__ float ssum[32];
  const int tid = threadIdx.x;
  const int wave = tid >> 5, lane = tid & 31;
  const int Lm = lane & 15, hi = lane >> 4;
  const int nBase = wave * 128;
  const size_t row0 = (size_t)blockIdx.x * 32;
  const int b = (int)(row0 >> 11);     // S = 2048, 32 | 2048 -> single batch per block

  if (tid < 32) ssum[tid] = 0.f;

  v8f acc[2][8] = {};

  for (int k0 = 0; k0 < K; k0 += KC) {
    for (int idx = tid; idx < 32 * (KC / 4); idx += 256) {
      const int row = idx >> 7;
      const int c4  = (idx & 127) << 2;
      const float* gp = context + (row0 + row) * (size_t)K + k0 + c4;
      if (k0 + KC < K) __builtin_prefetch(gp + KC, 0, 0);  // global_prefetch_b8
      float4 f = *(const float4*)gp;
      v4h h;
      h[0] = (_Float16)f.x; h[1] = (_Float16)f.y;
      h[2] = (_Float16)f.z; h[3] = (_Float16)f.w;
      *(v4h*)&Ash[row * LDK + c4] = h;
    }
    __syncthreads();

    for (int kc = 0; kc < KC; kc += 32) {
      const _Float16* ap = &Ash[Lm * LDK + kc + hi * 8];
      const v16h a0 = load_frag(ap);
      const v16h a1 = load_frag(ap + 16 * LDK);
      const _Float16* wp = Wc + (size_t)(nBase + Lm) * K + k0 + kc + hi * 8;
#pragma unroll
      for (int j = 0; j < 8; ++j) {
        const v16h bf = load_frag(wp + (size_t)j * 16 * K);
        acc[0][j] = WMMA_F16(a0, bf, acc[0][j]);
        acc[1][j] = WMMA_F16(a1, bf, acc[1][j]);
      }
    }
    __syncthreads();
  }

  // fused epilogue: tanh + dot with W_v, reduced over the 1024 e-columns
  float part[2][8] = {};
#pragma unroll
  for (int j = 0; j < 8; ++j) {
    const int e = nBase + j * 16 + Lm;
    const float t  = t_ws[b * 1024 + e];   // target + b_c, precomputed
    const float wv = Wv[e];
#pragma unroll
    for (int ms = 0; ms < 2; ++ms)
#pragma unroll
      for (int r = 0; r < 8; ++r)
        part[ms][r] += wv * tanhf(t + acc[ms][j][r]);
  }
  // reduce across the 16 N-lanes of each half-wave (xor bits 0..3 keep hi fixed)
#pragma unroll
  for (int mask = 1; mask < 16; mask <<= 1)
#pragma unroll
    for (int ms = 0; ms < 2; ++ms)
#pragma unroll
      for (int r = 0; r < 8; ++r)
        part[ms][r] += __shfl_xor(part[ms][r], mask, 32);

  if (Lm == 0) {
#pragma unroll
    for (int ms = 0; ms < 2; ++ms)
#pragma unroll
      for (int r = 0; r < 8; ++r)
        atomicAdd(&ssum[ms * 16 + hi * 8 + r], part[ms][r]);  // ds_add_f32
  }
  __syncthreads();
  if (tid < 32) scores[row0 + tid] = ssum[tid];
}

// ---------------------------------------------------------------------------
// Row softmax over S=2048, one block per batch.
// ---------------------------------------------------------------------------
__global__ __launch_bounds__(256) void softmax_kernel(const float* __restrict__ scores,
                                                      float* __restrict__ attn) {
  __shared__ float red[256];
  const int b = blockIdx.x, tid = threadIdx.x;
  const float* s = scores + b * 2048;
  float v[8], mx = -1e30f;
#pragma unroll
  for (int i = 0; i < 8; ++i) { v[i] = s[tid + i * 256]; mx = fmaxf(mx, v[i]); }
  red[tid] = mx; __syncthreads();
  for (int off = 128; off > 0; off >>= 1) {
    if (tid < off) red[tid] = fmaxf(red[tid], red[tid + off]);
    __syncthreads();
  }
  mx = red[0]; __syncthreads();
  float sum = 0.f;
#pragma unroll
  for (int i = 0; i < 8; ++i) { v[i] = expf(v[i] - mx); sum += v[i]; }
  red[tid] = sum; __syncthreads();
  for (int off = 128; off > 0; off >>= 1) {
    if (tid < off) red[tid] += red[tid + off];
    __syncthreads();
  }
  const float inv = 1.f / red[0];
#pragma unroll
  for (int i = 0; i < 8; ++i) attn[b * 2048 + tid + i * 256] = v[i] * inv;
}

// ---------------------------------------------------------------------------
// weighted_context[b,d] = sum_s attn[b,s] * context[b,s,d]   (streams context once)
// grid (32 batches, 8 d-segments) x 128 threads
// ---------------------------------------------------------------------------
__global__ __launch_bounds__(128) void wctx_kernel(const float* __restrict__ context,
                                                   const float* __restrict__ attn,
                                                   float* __restrict__ wc) {
  const int b = blockIdx.x;
  const int d = blockIdx.y * 128 + threadIdx.x;
  const float* cp = context + (size_t)b * 2048 * 1024 + d;
  const float* wp = attn + b * 2048;
  float acc = 0.f;
  for (int s = 0; s < 2048; s += 8) {
#pragma unroll
    for (int u = 0; u < 8; ++u)
      acc += wp[s + u] * cp[(size_t)(s + u) * 1024];
  }
  wc[b * 1024 + d] = acc;
}

// cat[m, 0:1024] = wc[m]; cat[m, 1024:2048] = input[m]
__global__ __launch_bounds__(256) void concat_kernel(const float* __restrict__ wc,
                                                     const float* __restrict__ input,
                                                     float* __restrict__ cat) {
  const int i = blockIdx.x * 256 + threadIdx.x;   // 65536 total
  const int m = i >> 11, c = i & 2047;
  cat[i] = (c < 1024) ? wc[m * 1024 + c] : input[m * 1024 + (c - 1024)];
}

// ---------------------------------------------------------------------------
extern "C" void kernel_launch(void* const* d_in, const int* in_sizes, int n_in,
                              void* d_out, int out_size, void* d_ws, size_t ws_size,
                              hipStream_t stream) {
  const float* input   = (const float*)d_in[0];   // (32, 1024)
  const float* context = (const float*)d_in[1];   // (32, 2048, 1024)
  const float* W_in    = (const float*)d_in[2];   // (1024, 1024)
  const float* W_c     = (const float*)d_in[3];   // (1024, 1024)
  const float* b_c     = (const float*)d_in[4];   // (1024,)
  const float* W_v     = (const float*)d_in[5];   // (1024,)
  const float* W_out   = (const float*)d_in[6];   // (1024, 2048)

  float* h_tilde = (float*)d_out;                 // (32,1024)
  float* wctx    = h_tilde + 32 * 1024;           // (32,1024)
  float* attn    = wctx + 32 * 1024;              // (32,2048)

  char* ws = (char*)d_ws;
  _Float16* Wc_h   = (_Float16*)(ws);                        // 2 MB
  _Float16* Win_h  = (_Float16*)(ws + (2u << 20));           // 2 MB
  _Float16* Wout_h = (_Float16*)(ws + (4u << 20));           // 4 MB
  float*    t_ws   = (float*)   (ws + (8u << 20));           // 128 KB (target + b_c)
  float*    scores = (float*)   (ws + (8u << 20) + (128u << 10)); // 256 KB
  float*    cat    = (float*)   (ws + (8u << 20) + (384u << 10)); // 256 KB

  // 1) weights fp32 -> f16 (live in L2 for all GEMMs)
  cvt_f16_kernel<<<(1 << 20) / 4 / 256, 256, 0, stream>>>(W_c,   Wc_h,   (1 << 20) / 4);
  cvt_f16_kernel<<<(1 << 20) / 4 / 256, 256, 0, stream>>>(W_in,  Win_h,  (1 << 20) / 4);
  cvt_f16_kernel<<<(2 << 20) / 4 / 256, 256, 0, stream>>>(W_out, Wout_h, (2 << 20) / 4);

  // 2) t_ws = input @ W_in.T + b_c
  gemm32_kernel<<<1, 256, 0, stream>>>(input, Win_h, b_c, t_ws, 1024, /*mode=*/0);

  // 3) fused additive-attention scores (the 137-GFLOP GEMM, never materializes source)
  attn_scores_kernel<<<2048, 256, 0, stream>>>(context, Wc_h, t_ws, W_v, scores);

  // 4) softmax over s
  softmax_kernel<<<32, 256, 0, stream>>>(scores, attn);

  // 5) weighted context (second and final streaming pass over context)
  wctx_kernel<<<dim3(32, 8), 128, 0, stream>>>(context, attn, wctx);

  // 6) concat + output projection with tanh
  concat_kernel<<<65536 / 256, 256, 0, stream>>>(wctx, input, cat);
  gemm32_kernel<<<1, 256, 0, stream>>>(cat, Wout_h, nullptr, h_tilde, 2048, /*mode=*/1);
}
